// RNN_76141180223667
// MI455X (gfx1250) — matrix operand order repaired and checked by reading, not compiled, not run on hardware
//
#include <hip/hip_runtime.h>

typedef __attribute__((ext_vector_type(16))) _Float16 v16h;
typedef __attribute__((ext_vector_type(8)))  float    v8f;
typedef __attribute__((ext_vector_type(4)))  unsigned int u32x4;
typedef __attribute__((ext_vector_type(8)))  int          i32x8;
typedef __attribute__((ext_vector_type(4)))  int          i32x4;

#define IN_N   57
#define HID    128
#define NCAT   18
#define BATCH  4096
#define TSTEPS 128
#define KPAD   64    // K=57 padded to 64 for the input projection
#define XBLK   1024  // halves per (tile,t) swizzled x block (2 KB)

#if __has_builtin(__builtin_amdgcn_tensor_load_to_lds)
#define HAVE_TDM 1
#else
#define HAVE_TDM 0
#endif
#if __has_include(<hip/amd_detail/amd_gfx1250_TDM.h>)
#define TDM_6ARG 1
#else
#define TDM_6ARG 0
#endif

__device__ __forceinline__ float sigmoid_f(float x) {
  return 1.0f / (1.0f + __expf(-x));
}
__device__ __forceinline__ float tanh_f(float x) {
  return 2.0f / (1.0f + __expf(-2.0f * x)) - 1.0f;
}

// B fragment: W row-major [N][K] f16. Lane n = n0+(lane&15); lanes 0-15 hold
// K = k0 + {0..7, 16..23}, lanes 16-31 hold K = k0 + {8..15, 24..31}.
__device__ __forceinline__ v16h load_b_frag(const _Float16* __restrict__ W,
                                            int n0, int k0, int ldk) {
  const int lane = threadIdx.x & 31;
  const _Float16* p =
      W + (size_t)(n0 + (lane & 15)) * ldk + k0 + ((lane >> 4) << 3);
  v16h r;
  reinterpret_cast<float4*>(&r)[0] = reinterpret_cast<const float4*>(p)[0];
  reinterpret_cast<float4*>(&r)[1] = reinterpret_cast<const float4*>(p + 16)[0];
  return r;
}

// A fragment from LDS h tile [16][HID] f16. Lane row m = lane&15; lanes 0-15:
// K = kt*32 + {0..7, 16..23}; lanes 16-31: K = kt*32 + {8..15, 24..31}.
__device__ __forceinline__ v16h load_a_lds(const _Float16* hs, int kt) {
  const int lane = threadIdx.x & 31;
  const _Float16* p = hs + (lane & 15) * HID + kt * 32 + ((lane >> 4) << 3);
  v16h r;
  reinterpret_cast<float4*>(&r)[0] = *reinterpret_cast<const float4*>(p);
  reinterpret_cast<float4*>(&r)[1] = *reinterpret_cast<const float4*>(p + 16);
  return r;
}

__device__ __forceinline__ v8f load_acc_lds(const float* p) {
  v8f r;
  reinterpret_cast<float4*>(&r)[0] = reinterpret_cast<const float4*>(p)[0];
  reinterpret_cast<float4*>(&r)[1] = reinterpret_cast<const float4*>(p)[1];
  return r;
}

// 1-D TDM copy of one 2 KB swizzled-x block (256 x 8B) into LDS.
__device__ __forceinline__ void tdm_load_block(const _Float16* __restrict__ g,
                                               void* l) {
#if HAVE_TDM
  const unsigned long long ga = (unsigned long long)(uintptr_t)g;
  const unsigned int lo = (unsigned int)(uintptr_t)l;  // LDS offset (addr[31:0])
  const unsigned int ndw = 256;                        // 8-byte elements
  u32x4 g0 = {1u,                                      // count=1, user D#
              lo,                                      // lds_addr
              (unsigned int)(ga & 0xffffffffu),        // global_addr[31:0]
              (unsigned int)((ga >> 32) & 0x01ffffffu) | (2u << 30)};  // type=2
  i32x8 g1 = {(int)(3u << 16),   // workgroup_mask=0, data_size=3 (8B)
              (int)(ndw << 16),  // tensor_dim0[15:0] in bits 63:48
              0,                 // tensor_dim0 hi / tensor_dim1 lo = 0
              (int)(ndw << 16),  // tile_dim0 in bits 127:112
              0,                 // tile_dim1 = tile_dim2 = 0 (unused -> 1-D)
              (int)ndw,          // tensor_dim0_stride[31:0]
              0, 0};
  i32x4 gz = {0, 0, 0, 0};
#if TDM_6ARG
  i32x8 gz8 = {0, 0, 0, 0, 0, 0, 0, 0};
  __builtin_amdgcn_tensor_load_to_lds(g0, g1, gz, gz, gz8, 0);
#else
  __builtin_amdgcn_tensor_load_to_lds(g0, g1, gz, gz, 0);
#endif
#else
  // Fallback: wave copies only the first 1 KB of the block with vector loads/stores.
  const int lane = threadIdx.x & 31;
  const float4* s = reinterpret_cast<const float4*>(g) + lane * 2;
  float4* d = reinterpret_cast<float4*>(l) + lane * 2;
  d[0] = s[0];
  d[1] = s[1];
#endif
}

// Pack weights to f16 (W_ih zero-padded K 57->64) and fold the two biases.
__global__ void lstm_prep_kernel(const float* __restrict__ W_ih,
                                 const float* __restrict__ W_hh,
                                 const float* __restrict__ b_ih,
                                 const float* __restrict__ b_hh,
                                 _Float16* __restrict__ wih,
                                 _Float16* __restrict__ whh,
                                 float* __restrict__ bias) {
  const int i = blockIdx.x * blockDim.x + threadIdx.x;
  if (i < 4 * HID * KPAD) {
    const int r = i / KPAD, k = i % KPAD;
    wih[i] = (k < IN_N) ? (_Float16)W_ih[r * IN_N + k] : (_Float16)0.0f;
  }
  if (i < 4 * HID * HID) whh[i] = (_Float16)W_hh[i];
  if (i < 4 * HID) bias[i] = b_ih[i] + b_hh[i];
}

// Swizzle x into per-(tile,t) A-fragment blocks: [tile][t][lane][32 halves],
// where each lane's 32 halves are its two v16h A-fragments (K 0..31, 32..63),
// f32->f16 converted and zero-padded for K >= 57. 2 KB per (tile,t).
__global__ __launch_bounds__(32) void lstm_xswizzle_kernel(
    const float* __restrict__ x, _Float16* __restrict__ xsw) {
  const int tile = blockIdx.x;  // 0..255
  const int t = blockIdx.y;     // 0..127
  const int lane = threadIdx.x & 31;
  const size_t base = ((size_t)(tile * 16 + (lane & 15)) * TSTEPS + t) * IN_N;
  _Float16 vals[32];
#pragma unroll
  for (int kt = 0; kt < 2; ++kt) {
    const int kb = kt * 32 + ((lane >> 4) << 3);
#pragma unroll
    for (int j = 0; j < 8; ++j) {
      const int k0 = kb + j, k1 = kb + 16 + j;
      vals[kt * 16 + j] = (k0 < IN_N) ? (_Float16)x[base + k0] : (_Float16)0.0f;
      vals[kt * 16 + 8 + j] =
          (k1 < IN_N) ? (_Float16)x[base + k1] : (_Float16)0.0f;
    }
  }
  _Float16* dst = xsw + (((size_t)tile * TSTEPS + t) * 32 + lane) * 32;
#pragma unroll
  for (int q = 0; q < 32; ++q) dst[q] = vals[q];
}

// One workgroup (4 waves) owns 16 batch rows for all 128 timesteps.
// Wave w computes gate w (i/f/g/o): 8 N-tiles x (2 x-WMMA + 4 h-WMMA);
// the compiler keeps all 48 f16 B-fragments resident in high VGPRs.
// x fragments arrive via TDM double-buffering into LDS; gate pre-activations
// are exchanged through LDS; c stays in registers; h staged as f16 in LDS.
__global__ __launch_bounds__(128, 1) void lstm_scan_kernel(
    const _Float16* __restrict__ xsw, const _Float16* __restrict__ wih,
    const _Float16* __restrict__ whh, const float* __restrict__ bias,
    const float* __restrict__ W_out, const float* __restrict__ b_out,
    float* __restrict__ out) {
  __shared__ float accs[4][8][32][8];              // [gate][jtile][lane][r] 32 KB
  __shared__ _Float16 hst[16 * HID];               // h tile, row-major       4 KB
  __shared__ alignas(16) _Float16 xbuf[2][XBLK];   // x double buffer         4 KB

  const int lane = threadIdx.x & 31;
  const int wave = threadIdx.x >> 5;  // gate id: 0=i 1=f 2=g 3=o
  const int l15 = lane & 15;
  const int half = lane >> 4;
  const int row0 = blockIdx.x * 16;  // batch rows [row0, row0+16)

  const _Float16* __restrict__ xtile =
      xsw + (size_t)blockIdx.x * TSTEPS * XBLK;

  float bb[8];
#pragma unroll
  for (int j = 0; j < 8; ++j) bb[j] = bias[wave * HID + j * 16 + l15];

  v16h hfrag[4] = {};  // h as A-fragments (K = 0..127), starts at zero
  v8f cc[2] = {};      // c for this wave's two hidden-column tiles

  if (wave == 0) tdm_load_block(xtile, &xbuf[0][0]);  // stage t = 0

  for (int t = 0; t < TSTEPS; ++t) {
    if (wave == 0) {
      // Stage t+1 (dummy re-stage of the last block on the final step keeps
      // the tensorcnt wait argument constant; buffer (t+1)&1 is unread then).
      const int tn = (t + 1 < TSTEPS) ? (t + 1) : (TSTEPS - 1);
      tdm_load_block(xtile + (size_t)tn * XBLK, &xbuf[(t + 1) & 1][0]);
#if HAVE_TDM
      __builtin_amdgcn_s_wait_tensorcnt(1);  // in-order: copy(t) now complete
#endif
    }
    __syncthreads();  // xbuf[t&1] visible to all waves

    const _Float16* xb = &xbuf[t & 1][lane * 32];
    const v16h ax0 = *reinterpret_cast<const v16h*>(xb);
    const v16h ax1 = *reinterpret_cast<const v16h*>(xb + 16);

#pragma unroll
    for (int j = 0; j < 8; ++j) {
      const int n0 = wave * HID + j * 16;
      const float bv = bb[j];
      v8f acc = {bv, bv, bv, bv, bv, bv, bv, bv};
      acc = __builtin_amdgcn_wmma_f32_16x16x32_f16(
          false, ax0, false, load_b_frag(wih, n0, 0, KPAD), (short)0, acc,
          false, false);
      acc = __builtin_amdgcn_wmma_f32_16x16x32_f16(
          false, ax1, false, load_b_frag(wih, n0, 32, KPAD), (short)0, acc,
          false, false);
#pragma unroll
      for (int kt = 0; kt < 4; ++kt)
        acc = __builtin_amdgcn_wmma_f32_16x16x32_f16(
            false, hfrag[kt], false, load_b_frag(whh, n0, kt * 32, HID),
            (short)0, acc, false, false);
      float* p = &accs[wave][j][lane][0];
      reinterpret_cast<float4*>(p)[0] = reinterpret_cast<float4*>(&acc)[0];
      reinterpret_cast<float4*>(p)[1] = reinterpret_cast<float4*>(&acc)[1];
    }
    __syncthreads();

    // Elementwise LSTM cell update; wave owns hidden-column tiles 2w, 2w+1.
#pragma unroll
    for (int jj = 0; jj < 2; ++jj) {
      const int j = wave * 2 + jj;
      const v8f gi = load_acc_lds(&accs[0][j][lane][0]);
      const v8f gf = load_acc_lds(&accs[1][j][lane][0]);
      const v8f gg = load_acc_lds(&accs[2][j][lane][0]);
      const v8f go = load_acc_lds(&accs[3][j][lane][0]);
#pragma unroll
      for (int r = 0; r < 8; ++r) {
        const float iv = sigmoid_f(gi[r]);
        const float fv = sigmoid_f(gf[r]);
        const float gv = tanh_f(gg[r]);
        const float ov = sigmoid_f(go[r]);
        const float cv = fv * cc[jj][r] + iv * gv;
        cc[jj][r] = cv;
        const float hv = ov * tanh_f(cv);
        // D layout element (m = r + 8*half, n = lane&15) -> h[m][col]
        hst[(r + half * 8) * HID + j * 16 + l15] = (_Float16)hv;
      }
    }
    __syncthreads();

#pragma unroll
    for (int kt = 0; kt < 4; ++kt) hfrag[kt] = load_a_lds(hst, kt);
  }

  // Output head: logits = h_T @ W_out^T + b_out  (16 x 18, K = 128)
  for (int p = threadIdx.x; p < 16 * NCAT; p += 128) {
    const int m = p / NCAT, cx = p % NCAT;
    float s = b_out[cx];
#pragma unroll 8
    for (int k = 0; k < HID; ++k)
      s += (float)hst[m * HID + k] * W_out[cx * HID + k];
    out[(size_t)(row0 + m) * NCAT + cx] = s;
  }
}

extern "C" void kernel_launch(void* const* d_in, const int* in_sizes, int n_in,
                              void* d_out, int out_size, void* d_ws,
                              size_t ws_size, hipStream_t stream) {
  (void)in_sizes; (void)n_in; (void)out_size; (void)ws_size;
  const float* x     = (const float*)d_in[0];
  const float* W_ih  = (const float*)d_in[1];
  const float* W_hh  = (const float*)d_in[2];
  const float* b_ih  = (const float*)d_in[3];
  const float* b_hh  = (const float*)d_in[4];
  const float* W_out = (const float*)d_in[5];
  const float* b_out = (const float*)d_in[6];
  float* out = (float*)d_out;

  char* ws = (char*)d_ws;
  _Float16* wih  = (_Float16*)ws;                   // 512*64  f16 =  64 KB
  _Float16* whh  = (_Float16*)(ws + (64 << 10));    // 512*128 f16 = 128 KB
  float*    bias = (float*)(ws + (192 << 10));      // 512 f32     =   2 KB
  _Float16* xsw  = (_Float16*)(ws + (256 << 10));   // 256*128*2KB =  64 MB

  lstm_prep_kernel<<<256, 256, 0, stream>>>(W_ih, W_hh, b_ih, b_hh, wih, whh,
                                            bias);
  lstm_xswizzle_kernel<<<dim3(BATCH / 16, TSTEPS), 32, 0, stream>>>(x, xsw);
  lstm_scan_kernel<<<BATCH / 16, 128, 0, stream>>>(xsw, wih, whh, bias, W_out,
                                                   b_out, out);
}
